// C2FTopDownCore_68934225101258
// MI455X (gfx1250) — matrix operand order repaired and checked by reading, not compiled, not run on hardware
//
#include <hip/hip_runtime.h>
#include <hip/hip_bf16.h>
#include <stddef.h>

// ---------------------------------------------------------------------------
// C2F TopDown core for MI455X (gfx1250, wave32, WMMA).
// All GEMMs: fp32 inputs converted to bf16 in LDS, v_wmma_f32_16x16x32_bf16.
// ---------------------------------------------------------------------------

#define BATCH 128
#define RDIM  1024
#define ADIM  512
#define FDIM  2048
#define LLEN  196

typedef __bf16        bf16x16 __attribute__((ext_vector_type(16)));
typedef __bf16        bf16x2  __attribute__((ext_vector_type(2)));
typedef float         f32x8   __attribute__((ext_vector_type(8)));
typedef float         f32x4   __attribute__((ext_vector_type(4)));
typedef float         f32x2   __attribute__((ext_vector_type(2)));
typedef unsigned int  u32x4   __attribute__((ext_vector_type(4)));
typedef unsigned int  u32x2   __attribute__((ext_vector_type(2)));

// fp32 -> packed bf16 pair; plain casts let the compiler pick the native
// CDNA5 packed convert instead of a manual integer-RNE sequence.
__device__ __forceinline__ unsigned int pack2bf(float a, float b) {
  bf16x2 p;
  p.x = (__bf16)a;
  p.y = (__bf16)b;
  return __builtin_bit_cast(unsigned int, p);
}
__device__ __forceinline__ float sigf(float x) { return 1.f / (1.f + __expf(-x)); }

// ---------------------------------------------------------------------------
// Generic multi-segment GEMM: C[128,N] = sum_s A_s[128,K_s] * W_s[N,K_s]^T
//                                        + b0[N] + b1[N]
// Block tile 128(M) x 64(N), K-step 32. 8 waves; wave w owns rows
// [16w,16w+16) x 64 cols = 4 WMMA accumulators.
// Outer loop over segments (all fields loop-invariant registers), inner loop
// software-pipelined: global loads of tile k+1 overlap WMMAs of tile k.
// ---------------------------------------------------------------------------
struct Seg { const float* A; const float* W; int ldw; int wofs; int klen; };
struct GemmArgs {
  Seg seg[4];
  int nseg;
  const float* b0;
  const float* b1;
  float* C;
  int N;
};

#define GBK  32
#define LDSP 40   // padded LDS row stride in bf16 elems (80B: 16B-aligned, conflict-spread)

struct TileRegs { f32x4 a[4]; f32x4 b[2]; };

__device__ __forceinline__ void load_tile(TileRegs& r, const float* Ap, int lda, int kloc,
                                          const float* Wp, int ldw, int wcol,
                                          int nBase, int tid) {
#pragma unroll
  for (int i = 0; i < 4; ++i) {
    const int e = (tid + i * 256) * 4, row = e >> 5, kk = e & 31;
    r.a[i] = *reinterpret_cast<const f32x4*>(Ap + (size_t)row * lda + kloc + kk);
  }
#pragma unroll
  for (int i = 0; i < 2; ++i) {
    const int e = (tid + i * 256) * 4, row = e >> 5, kk = e & 31;
    r.b[i] = *reinterpret_cast<const f32x4*>(Wp + (size_t)(nBase + row) * ldw + wcol + kk);
  }
}

__device__ __forceinline__ void store_tile(const TileRegs& r, unsigned short* la,
                                           unsigned short* lb, int tid) {
#pragma unroll
  for (int i = 0; i < 4; ++i) {
    const int e = (tid + i * 256) * 4, row = e >> 5, kk = e & 31;
    u32x2 p; p.x = pack2bf(r.a[i].x, r.a[i].y); p.y = pack2bf(r.a[i].z, r.a[i].w);
    *reinterpret_cast<u32x2*>(&la[row * LDSP + kk]) = p;   // ds_store_b64
  }
#pragma unroll
  for (int i = 0; i < 2; ++i) {
    const int e = (tid + i * 256) * 4, row = e >> 5, kk = e & 31;
    u32x2 p; p.x = pack2bf(r.b[i].x, r.b[i].y); p.y = pack2bf(r.b[i].z, r.b[i].w);
    *reinterpret_cast<u32x2*>(&lb[row * LDSP + kk]) = p;
  }
}

__global__ __launch_bounds__(256, 2)
void gemm_bf16_wmma(GemmArgs g) {
  __shared__ unsigned short la[128 * LDSP];   // A tile 128 x 32 (bf16)
  __shared__ unsigned short lb[64 * LDSP];    // B tile 64  x 32 (bf16) == W rows

  const int tid   = threadIdx.x;
  const int wave  = tid >> 5;
  const int lane  = tid & 31;
  const int hl    = lane >> 4;   // lane half (K-group select per ISA 7.12.2)
  const int l16   = lane & 15;
  const int nBase = blockIdx.x * 64;

  f32x8 acc[4];
#pragma unroll
  for (int t = 0; t < 4; ++t) {
    const int n = nBase + t * 16 + l16;
    float bv = 0.f;
    if (g.b0) bv += g.b0[n];
    if (g.b1) bv += g.b1[n];
#pragma unroll
    for (int r = 0; r < 8; ++r) acc[t][r] = bv;
  }

  // fragment LDS base pointers (invariant)
  const unsigned short* ap = &la[(wave * 16 + l16) * LDSP + hl * 8];
  const unsigned short* bp0 = &lb[l16 * LDSP + hl * 8];

  const int nseg = g.nseg;
  for (int s = 0; s < nseg; ++s) {
    const float* Ap  = g.seg[s].A;
    const float* Wp  = g.seg[s].W;
    const int    ldw = g.seg[s].ldw;
    const int    wof = g.seg[s].wofs;
    const int    lda = g.seg[s].klen;
    const int    ksteps = lda / GBK;

    TileRegs r;
    load_tile(r, Ap, lda, 0, Wp, ldw, wof, nBase, tid);

    for (int ks = 0; ks < ksteps; ++ks) {
      store_tile(r, la, lb, tid);
      __syncthreads();                    // tile ks visible in LDS

      if (ks + 1 < ksteps)                // prefetch tile ks+1 under the WMMAs
        load_tile(r, Ap, lda, (ks + 1) * GBK, Wp, ldw, wof + (ks + 1) * GBK, nBase, tid);

      // Fragments per ISA 7.12.2 (wave32):
      //  A 16x32: lane -> row M=l16; K = {8hl..8hl+7} U {16+8hl..16+8hl+7}
      //  B 32x16: lane -> col N=l16; K = {8hl..8hl+7} U {16+8hl..16+8hl+7}
      union Frag { bf16x16 v; u32x4 q[2]; };
      Frag a, bb[4];
      a.q[0] = *reinterpret_cast<const u32x4*>(ap);        // ds_load_b128 x2
      a.q[1] = *reinterpret_cast<const u32x4*>(ap + 16);
#pragma unroll
      for (int t = 0; t < 4; ++t) {
        const unsigned short* bp = bp0 + t * 16 * LDSP;
        bb[t].q[0] = *reinterpret_cast<const u32x4*>(bp);
        bb[t].q[1] = *reinterpret_cast<const u32x4*>(bp + 16);
      }
#pragma unroll
      for (int t = 0; t < 4; ++t)
        acc[t] = __builtin_amdgcn_wmma_f32_16x16x32_bf16(
            /*neg_a=*/false, a.v, /*neg_b=*/false, bb[t].v,
            /*c_mod=*/(short)0, acc[t], /*reuse_a=*/false, /*reuse_b=*/false);

      __syncthreads();                    // LDS reads done before next store
    }
  }

  // store: C/D layout -> lane owns (M = 16*wave + 8*hl + r, N = nBase+16t+l16)
#pragma unroll
  for (int t = 0; t < 4; ++t) {
    const int n = nBase + t * 16 + l16;
    float* cp = g.C + (size_t)(wave * 16 + hl * 8) * g.N + n;
#pragma unroll
    for (int r = 0; r < 8; ++r) cp[(size_t)r * g.N] = acc[t][r];
  }
}

// ---------------------------------------------------------------------------
// LSTM pointwise: gates[B,4R] + c_in[B,R] -> h_out, c_out
// ---------------------------------------------------------------------------
__global__ __launch_bounds__(256)
void lstm_pw(const float* __restrict__ gates, const float* __restrict__ c_in,
             float* __restrict__ h_out, float* __restrict__ c_out) {
  const int idx = blockIdx.x * 256 + threadIdx.x;      // < B*R
  const int m = idx >> 10, r = idx & (RDIM - 1);
  const float* grow = gates + (size_t)m * 4 * RDIM;
  const float gi = grow[r];
  const float gf = grow[RDIM + r];
  const float gg = grow[2 * RDIM + r];
  const float go = grow[3 * RDIM + r];
  const float c  = c_in[idx];
  const float cn = sigf(gf) * c + sigf(gi) * tanhf(gg);
  c_out[idx] = cn;
  h_out[idx] = sigf(go) * tanhf(cn);
}

// ---------------------------------------------------------------------------
// GLU: raw[B,2R] -> out[B,R] = tanh(raw[:, :R]) * sigmoid(raw[:, R:])
// ---------------------------------------------------------------------------
__global__ __launch_bounds__(256)
void glu_pw(const float* __restrict__ raw, float* __restrict__ out) {
  const int idx = blockIdx.x * 256 + threadIdx.x;
  const int m = idx >> 10, r = idx & (RDIM - 1);
  const float a = raw[(size_t)m * 2 * RDIM + r];
  const float b = raw[(size_t)m * 2 * RDIM + RDIM + r];
  out[idx] = tanhf(a) * sigf(b);
}

// out[B,R] = tanh(raw[:, :R] + add) * sigmoid(raw[:, R:] + add)
__global__ __launch_bounds__(256)
void outact_pw(const float* __restrict__ raw, const float* __restrict__ add,
               float* __restrict__ out) {
  const int idx = blockIdx.x * 256 + threadIdx.x;
  const int m = idx >> 10, r = idx & (RDIM - 1);
  const float x = add[idx];
  const float a = raw[(size_t)m * 2 * RDIM + r] + x;
  const float b = raw[(size_t)m * 2 * RDIM + RDIM + r] + x;
  out[idx] = tanhf(a) * sigf(b);
}

__global__ __launch_bounds__(256)
void add_pw(const float* __restrict__ a, const float* __restrict__ b,
            float* __restrict__ out) {
  const int idx = blockIdx.x * 256 + threadIdx.x;
  out[idx] = a[idx] + b[idx];
}

// ---------------------------------------------------------------------------
// Attention scores + softmax: one block per batch row.
//   score[l] = sum_a tanh(p[l,a]+h[a]) * sig(p[l,a+A]+h[a+A]) * Wa[a] + ba
//   w = softmax(score)
// Streams p_att (103 MB) exactly once.
// ---------------------------------------------------------------------------
__global__ __launch_bounds__(256)
void att_scores(const float* __restrict__ p_att, const float* __restrict__ atth,
                const float* __restrict__ Wa, const float* __restrict__ ba,
                float* __restrict__ wout) {
  __shared__ float sh[2 * ADIM];
  __shared__ float ssc[256];
  __shared__ float red[256];
  const int b = blockIdx.x;
  const int tid = threadIdx.x;
  for (int i = tid; i < 2 * ADIM; i += 256) sh[i] = atth[(size_t)b * 2 * ADIM + i];
  __syncthreads();

  const int wave = tid >> 5, lane = tid & 31;
  for (int l = wave; l < LLEN; l += 8) {
    const float* pr = p_att + ((size_t)b * LLEN + l) * (2 * ADIM);
    float sum = 0.f;
    for (int a = lane; a < ADIM; a += 32) {
      const float t  = tanhf(pr[a] + sh[a]);
      const float gt = sigf(pr[a + ADIM] + sh[a + ADIM]);
      sum += t * gt * Wa[a];
    }
#pragma unroll
    for (int off = 16; off > 0; off >>= 1) sum += __shfl_down(sum, off, 32);
    if (lane == 0) ssc[l] = sum + ba[0];
  }
  __syncthreads();

  // softmax over LLEN
  float m = -1e30f;
  for (int i = tid; i < LLEN; i += 256) m = fmaxf(m, ssc[i]);
  red[tid] = m; __syncthreads();
  for (int s = 128; s > 0; s >>= 1) {
    if (tid < s) red[tid] = fmaxf(red[tid], red[tid + s]);
    __syncthreads();
  }
  const float mx = red[0];
  __syncthreads();
  float sum = 0.f;
  for (int i = tid; i < LLEN; i += 256) { const float e = __expf(ssc[i] - mx); ssc[i] = e; sum += e; }
  red[tid] = sum; __syncthreads();
  for (int s = 128; s > 0; s >>= 1) {
    if (tid < s) red[tid] += red[tid + s];
    __syncthreads();
  }
  const float inv = 1.f / red[0];
  for (int i = tid; i < LLEN; i += 256) wout[(size_t)b * LLEN + i] = ssc[i] * inv;
}

// ---------------------------------------------------------------------------
// Weighted feature sum: out[b,f] = sum_l w[b,l] * feats[b,l,f]
// grid (B, F/512); streams att_feats (205 MB) with float2 loads + prefetch.
// ---------------------------------------------------------------------------
__global__ __launch_bounds__(256)
void att_wsum(const float* __restrict__ w, const float* __restrict__ feats,
              float* __restrict__ out) {
  __shared__ float sw[LLEN];
  const int b = blockIdx.x;
  for (int i = threadIdx.x; i < LLEN; i += 256) sw[i] = w[(size_t)b * LLEN + i];
  __syncthreads();
  const int f = blockIdx.y * 512 + threadIdx.x * 2;
  const float* fp = feats + (size_t)b * LLEN * FDIM + f;
  f32x2 acc; acc.x = 0.f; acc.y = 0.f;
  for (int l = 0; l < LLEN; ++l) {
    __builtin_prefetch(fp + (size_t)(l + 6) * FDIM, 0, 1);  // global_prefetch_b8
    const f32x2 v = *reinterpret_cast<const f32x2*>(fp + (size_t)l * FDIM);
    acc.x += sw[l] * v.x;
    acc.y += sw[l] * v.y;
  }
  *reinterpret_cast<f32x2*>(out + (size_t)b * FDIM + f) = acc;
}

// ---------------------------------------------------------------------------
// Host-side orchestration
// ---------------------------------------------------------------------------
static inline void launch_gemm(hipStream_t st, const GemmArgs& a) {
  gemm_bf16_wmma<<<dim3(a.N / 64), 256, 0, st>>>(a);
}
static inline Seg mkseg(const float* A, const float* W, int ldw, int wofs, int klen) {
  Seg s; s.A = A; s.W = W; s.ldw = ldw; s.wofs = wofs; s.klen = klen; return s;
}

extern "C" void kernel_launch(void* const* d_in, const int* in_sizes, int n_in,
                              void* d_out, int out_size, void* d_ws, size_t ws_size,
                              hipStream_t stream) {
  (void)in_sizes; (void)n_in; (void)out_size; (void)ws_size;
  // ---- inputs (setup_inputs dict order, params flattened in insertion order)
  const float* xt          = (const float*)d_in[0];
  const float* xt_fine     = (const float*)d_in[1];
  const float* fc_feats    = (const float*)d_in[2];
  const float* att_feats   = (const float*)d_in[3];
  const float* p_att       = (const float*)d_in[4];
  const float* p_att_final = (const float*)d_in[5];
  const float* state_h     = (const float*)d_in[6];   // [3,B,R]
  const float* state_c     = (const float*)d_in[7];   // [3,B,R]
  const float* Wih_att = (const float*)d_in[8],  *Whh_att = (const float*)d_in[9];
  const float* bih_att = (const float*)d_in[10], *bhh_att = (const float*)d_in[11];
  const float* Wih_f   = (const float*)d_in[12], *Whh_f   = (const float*)d_in[13];
  const float* bih_f   = (const float*)d_in[14], *bhh_f   = (const float*)d_in[15];
  const float* Wih_l   = (const float*)d_in[16], *Whh_l   = (const float*)d_in[17];
  const float* bih_l   = (const float*)d_in[18], *bhh_l   = (const float*)d_in[19];
  const float* Wh2a_f  = (const float*)d_in[20], *bh2a_f  = (const float*)d_in[21];
  const float* Wa_f    = (const float*)d_in[22], *ba_f    = (const float*)d_in[23];
  const float* Wao_f   = (const float*)d_in[24], *bao_f   = (const float*)d_in[25];
  const float* Wh2a_l  = (const float*)d_in[26], *bh2a_l  = (const float*)d_in[27];
  const float* Wa_l    = (const float*)d_in[28], *ba_l    = (const float*)d_in[29];
  const float* Wao_l   = (const float*)d_in[30], *bao_l   = (const float*)d_in[31];
  const float* Wpf     = (const float*)d_in[32], *bpf     = (const float*)d_in[33];
  const float* Wpl     = (const float*)d_in[34], *bpl     = (const float*)d_in[35];
  const float* Whf     = (const float*)d_in[36], *bhf     = (const float*)d_in[37];
  const float* Whl     = (const float*)d_in[38], *bhl     = (const float*)d_in[39];

  const int BR = BATCH * RDIM;          // 131072
  // ---- outputs: [out_fine | out_final | new_h(3) | new_c(3)]
  float* out_fine  = (float*)d_out;
  float* out_final = out_fine + BR;
  float* new_h     = out_final + BR;
  float* new_c     = new_h + 3 * BR;
  float* h_att = new_h;               float* c_att = new_c;
  float* h_lf  = new_h + BR;          float* c_lf  = new_c + BR;
  float* h_ll  = new_h + 2 * BR;      float* c_ll  = new_c + 2 * BR;

  // ---- workspace layout (floats)
  float* ws     = (float*)d_ws;
  float* gates  = ws;                    // 128*4096
  float* atth   = gates  + 128 * 4096;   // 128*1024
  float* attw   = atth   + 128 * 1024;   // 128*196
  float* attres = attw   + 128 * 256;    // 128*2048
  float* attraw = attres + 128 * 2048;   // 128*2048
  float* attact = attraw + 128 * 2048;   // 128*1024 (att_fine then att_final)
  float* hsum   = attact + 128 * 1024;   // 128*1024
  float* oraw   = hsum   + 128 * 1024;   // 128*2048

  const dim3 blk(256);
  const dim3 gridBR(BR / 256);           // 512 blocks for B*R pointwise

  GemmArgs a;

  // ===== 1. att_lstm gates: [prev_h | fc | xt] * Wih^T + h0 * Whh^T + biases
  a.seg[0] = mkseg(state_h + 2 * BR, Wih_att, 3 * RDIM, 0,        RDIM);
  a.seg[1] = mkseg(fc_feats,         Wih_att, 3 * RDIM, RDIM,     RDIM);
  a.seg[2] = mkseg(xt,               Wih_att, 3 * RDIM, 2 * RDIM, RDIM);
  a.seg[3] = mkseg(state_h,          Whh_att, RDIM,     0,        RDIM);
  a.nseg = 4; a.b0 = bih_att; a.b1 = bhh_att; a.C = gates; a.N = 4 * RDIM;
  launch_gemm(stream, a);
  lstm_pw<<<gridBR, blk, 0, stream>>>(gates, state_c, h_att, c_att);

  // ===== 2. fine attention
  a.seg[0] = mkseg(h_att, Wh2a_f, RDIM, 0, RDIM);
  a.nseg = 1; a.b0 = bh2a_f; a.b1 = nullptr; a.C = atth; a.N = 2 * ADIM;
  launch_gemm(stream, a);
  att_scores<<<dim3(BATCH), blk, 0, stream>>>(p_att, atth, Wa_f, ba_f, attw);
  att_wsum<<<dim3(BATCH, FDIM / 512), blk, 0, stream>>>(attw, att_feats, attres);
  a.seg[0] = mkseg(attres, Wao_f, FDIM, 0, FDIM);
  a.nseg = 1; a.b0 = bao_f; a.b1 = nullptr; a.C = attraw; a.N = 2 * RDIM;
  launch_gemm(stream, a);
  glu_pw<<<gridBR, blk, 0, stream>>>(attraw, attact);            // att_fine

  // ===== 3. fine_lstm gates: [att_fine | h_att] * Wih^T + h1 * Whh^T
  a.seg[0] = mkseg(attact,       Wih_f, 2 * RDIM, 0,    RDIM);
  a.seg[1] = mkseg(h_att,        Wih_f, 2 * RDIM, RDIM, RDIM);
  a.seg[2] = mkseg(state_h + BR, Whh_f, RDIM,     0,    RDIM);
  a.nseg = 3; a.b0 = bih_f; a.b1 = bhh_f; a.C = gates; a.N = 4 * RDIM;
  launch_gemm(stream, a);
  lstm_pw<<<gridBR, blk, 0, stream>>>(gates, state_c + BR, h_lf, c_lf);

  // ===== 4. output_fine: of = h_lf*Whf^T + bhf + att_fine*Wpf^T + bpf
  a.seg[0] = mkseg(h_lf,   Whf, RDIM, 0, RDIM);
  a.seg[1] = mkseg(attact, Wpf, RDIM, 0, RDIM);
  a.nseg = 2; a.b0 = bhf; a.b1 = bpf; a.C = oraw; a.N = 2 * RDIM;
  launch_gemm(stream, a);
  outact_pw<<<gridBR, blk, 0, stream>>>(oraw, xt_fine, out_fine);

  // ===== 5. final attention on (h_lf + att_fine)
  add_pw<<<gridBR, blk, 0, stream>>>(h_lf, attact, hsum);
  a.seg[0] = mkseg(hsum, Wh2a_l, RDIM, 0, RDIM);
  a.nseg = 1; a.b0 = bh2a_l; a.b1 = nullptr; a.C = atth; a.N = 2 * ADIM;
  launch_gemm(stream, a);
  att_scores<<<dim3(BATCH), blk, 0, stream>>>(p_att_final, atth, Wa_l, ba_l, attw);
  att_wsum<<<dim3(BATCH, FDIM / 512), blk, 0, stream>>>(attw, att_feats, attres);
  a.seg[0] = mkseg(attres, Wao_l, FDIM, 0, FDIM);
  a.nseg = 1; a.b0 = bao_l; a.b1 = nullptr; a.C = attraw; a.N = 2 * RDIM;
  launch_gemm(stream, a);
  glu_pw<<<gridBR, blk, 0, stream>>>(attraw, attact);            // att_final (att_fine dead)

  // ===== 6. final_lstm gates: [att_final | h_lf] * Wih^T + h2 * Whh^T
  a.seg[0] = mkseg(attact,           Wih_l, 2 * RDIM, 0,    RDIM);
  a.seg[1] = mkseg(h_lf,             Wih_l, 2 * RDIM, RDIM, RDIM);
  a.seg[2] = mkseg(state_h + 2 * BR, Whh_l, RDIM,     0,    RDIM);
  a.nseg = 3; a.b0 = bih_l; a.b1 = bhh_l; a.C = gates; a.N = 4 * RDIM;
  launch_gemm(stream, a);
  lstm_pw<<<gridBR, blk, 0, stream>>>(gates, state_c + 2 * BR, h_ll, c_ll);

  // ===== 7. output_final: ol = h_ll*Whl^T + bhl + att_final*Wpl^T + bpl
  a.seg[0] = mkseg(h_ll,   Whl, RDIM, 0, RDIM);
  a.seg[1] = mkseg(attact, Wpl, RDIM, 0, RDIM);
  a.nseg = 2; a.b0 = bhl; a.b1 = bpl; a.C = oraw; a.N = 2 * RDIM;
  launch_gemm(stream, a);
  outact_pw<<<gridBR, blk, 0, stream>>>(oraw, xt, out_final);
}
